// KMeans_26938034880803
// MI455X (gfx1250) — matrix operand order repaired and checked by reading, not compiled, not run on hardware
//
#include <hip/hip_runtime.h>
#include <hip/hip_bf16.h>

// ---------------------------------------------------------------------------
// KMeans on MI455X (gfx1250): assignment via V_WMMA_F32_16X16X32_BF16 with
// split-bf16 (hi+lo) inputs for ~f32 accuracy, bias (-0.5*||c||^2) folded into
// the WMMA accumulator, 2 point-tiles per wave sharing each B load,
// LDS-transposed in-lane argmax reduction, LDS-aggregated scatter-mean,
// persistent workgroups.
// ---------------------------------------------------------------------------

typedef __attribute__((ext_vector_type(16))) __bf16 v16bf;
typedef __attribute__((ext_vector_type(8)))  float  v8f;

#define KCENT 256
#define DIM   32
#define WAVES_PER_WG 8
#define WG_SIZE 256
#define PTS_PER_WAVE 32                     // two 16-point WMMA A tiles
#define KEY_ROW 17                          // u64 per point row (pad: bank-safe)

// dynamic LDS layout (bytes)
#define OFF_SUMS  0                         // float [256*32]  = 32768
#define OFF_CNT   32768                     // float [256]     = 1024
#define OFF_BIAS  33792                     // float [256]     = 1024
#define OFF_CHI   34816                     // __bf16[256*32]  = 16384
#define OFF_CLO   51200                     // __bf16[256*32]  = 16384
#define OFF_KEYS  67584                     // u64 [8][32][17] = 34816
#define SMEM_BYTES 102400

// monotone order-preserving f32 -> u32 map (for branchless argmax keys)
__device__ __forceinline__ unsigned int f32_ordered(float f) {
    unsigned int u = __float_as_uint(f);
    return (u & 0x80000000u) ? ~u : (u | 0x80000000u);
}

template<bool OUTPUT_MODE>
__global__ __launch_bounds__(WG_SIZE)
void kmeans_assign_kernel(const float* __restrict__ x,
                          const float* __restrict__ c,      // [256][32]
                          float* __restrict__ g_sums,       // [256][32]
                          float* __restrict__ g_cnt,        // [256]
                          float* __restrict__ out,          // [N][32]
                          int N, int numTiles)
{
    extern __shared__ char smem[];
    float*  s_sums = (float*)(smem + OFF_SUMS);
    float*  s_cnt  = (float*)(smem + OFF_CNT);
    float*  s_bias = (float*)(smem + OFF_BIAS);
    __bf16* s_chi  = (__bf16*)(smem + OFF_CHI);
    __bf16* s_clo  = (__bf16*)(smem + OFF_CLO);

    const int tid = threadIdx.x;

    // ---- stage centroids: hi/lo bf16 split + bias = -0.5*||c||^2 (f32) ----
    {
        float sq = 0.f;
        #pragma unroll 4
        for (int d = 0; d < DIM; ++d) {
            float v = c[tid * DIM + d];
            __bf16 hi = (__bf16)v;
            __bf16 lo = (__bf16)(v - (float)hi);
            s_chi[tid * DIM + d] = hi;
            s_clo[tid * DIM + d] = lo;
            sq += v * v;
            if (!OUTPUT_MODE) s_sums[tid * DIM + d] = 0.f;
        }
        s_bias[tid] = -0.5f * sq;
        if (!OUTPUT_MODE) s_cnt[tid] = 0.f;
    }
    __syncthreads();

    const int lane = tid & 31;
    const int wave = tid >> 5;
    const int grp  = lane >> 4;   // lane half: selects which dims this lane owns
    const int l16  = lane & 15;

    // per-wave argmax transpose scratch: [32 points][KEY_ROW u64]
    unsigned long long* wk =
        (unsigned long long*)(smem + OFF_KEYS) + (size_t)wave * 32 * KEY_ROW;

    const int tile0      = blockIdx.x * WAVES_PER_WG + wave;
    const int tileStride = gridDim.x * WAVES_PER_WG;

    for (int tile = tile0; tile < numTiles; tile += tileStride) {
        // prefetch next grid-stride tile toward L2 (global_prefetch_b8)
        {
            int ntile = tile + tileStride;
            if (ntile < numTiles) {
                const float* nx = x + (size_t)(ntile * PTS_PER_WAVE + l16) * DIM;
                __builtin_prefetch(nx, 0, 0);
                __builtin_prefetch(nx + 16 * DIM, 0, 0);
            }
        }

        // ------------- load two A tiles: 32 points x 32 dims ---------------
        // A layout (16-bit, 16x32): lanes 0-15 = point l16, dims {0-7,16-23};
        //                           lanes 16-31 = point l16, dims {8-15,24-31}
        int p1 = tile * PTS_PER_WAVE + l16;
        int p2 = p1 + 16;
        bool v1 = p1 < N, v2 = p2 < N;
        const float* xp1 = x + (size_t)(v1 ? p1 : (N - 1)) * DIM + 8 * grp;
        const float* xp2 = x + (size_t)(v2 ? p2 : (N - 1)) * DIM + 8 * grp;
        float af1[16], af2[16];
        {
            float4 a0 = *(const float4*)(xp1 + 0);
            float4 a1 = *(const float4*)(xp1 + 4);
            float4 a2 = *(const float4*)(xp1 + 16);
            float4 a3 = *(const float4*)(xp1 + 20);
            af1[0]=a0.x; af1[1]=a0.y; af1[2]=a0.z; af1[3]=a0.w;
            af1[4]=a1.x; af1[5]=a1.y; af1[6]=a1.z; af1[7]=a1.w;
            af1[8]=a2.x; af1[9]=a2.y; af1[10]=a2.z; af1[11]=a2.w;
            af1[12]=a3.x; af1[13]=a3.y; af1[14]=a3.z; af1[15]=a3.w;
            float4 b0 = *(const float4*)(xp2 + 0);
            float4 b1 = *(const float4*)(xp2 + 4);
            float4 b2 = *(const float4*)(xp2 + 16);
            float4 b3 = *(const float4*)(xp2 + 20);
            af2[0]=b0.x; af2[1]=b0.y; af2[2]=b0.z; af2[3]=b0.w;
            af2[4]=b1.x; af2[5]=b1.y; af2[6]=b1.z; af2[7]=b1.w;
            af2[8]=b2.x; af2[9]=b2.y; af2[10]=b2.z; af2[11]=b2.w;
            af2[12]=b3.x; af2[13]=b3.y; af2[14]=b3.z; af2[15]=b3.w;
        }
        v16bf ahi1, alo1, ahi2, alo2;
        #pragma unroll
        for (int e = 0; e < 16; ++e) {
            __bf16 h1 = (__bf16)af1[e];
            ahi1[e] = h1; alo1[e] = (__bf16)(af1[e] - (float)h1);
            __bf16 h2 = (__bf16)af2[e];
            ahi2[e] = h2; alo2[e] = (__bf16)(af2[e] - (float)h2);
        }

        // ------------- score all 256 centroids, track running argmax -------
        float bv1[8], bv2[8];
        int   bi1[8], bi2[8];
        #pragma unroll
        for (int r = 0; r < 8; ++r) {
            bv1[r] = -3.4e38f; bi1[r] = 0;
            bv2[r] = -3.4e38f; bi2[r] = 0;
        }

        #pragma unroll 2
        for (int ct = 0; ct < 16; ++ct) {
            // B layout (32x16): lanes 0-15 = centroid n, K={0-7,16-23};
            //                   lanes 16-31 = centroid n, K={8-15,24-31}
            int n = ct * 16 + l16;
            typedef __attribute__((ext_vector_type(8))) __bf16 v8bf_;
            v16bf bhi = __builtin_shufflevector(*(const v8bf_*)(s_chi + n * DIM + 8 * grp), *(const v8bf_*)(s_chi + n * DIM + 16 + 8 * grp), 0,1,2,3,4,5,6,7,8,9,10,11,12,13,14,15);
            v16bf blo = __builtin_shufflevector(*(const v8bf_*)(s_clo + n * DIM + 8 * grp), *(const v8bf_*)(s_clo + n * DIM + 16 + 8 * grp), 0,1,2,3,4,5,6,7,8,9,10,11,12,13,14,15);
            float bias = s_bias[n];
            v8f acc1, acc2;
            #pragma unroll
            for (int r = 0; r < 8; ++r) { acc1[r] = bias; acc2[r] = bias; }

            // score = xh*ch + xh*cl + xl*ch - 0.5*||c||^2  (f32 accumulate)
            acc1 = __builtin_amdgcn_wmma_f32_16x16x32_bf16(
                       false, ahi1, false, bhi, (short)0, acc1, false, false);
            acc2 = __builtin_amdgcn_wmma_f32_16x16x32_bf16(
                       false, ahi2, false, bhi, (short)0, acc2, false, false);
            acc1 = __builtin_amdgcn_wmma_f32_16x16x32_bf16(
                       false, ahi1, false, blo, (short)0, acc1, false, false);
            acc2 = __builtin_amdgcn_wmma_f32_16x16x32_bf16(
                       false, ahi2, false, blo, (short)0, acc2, false, false);
            acc1 = __builtin_amdgcn_wmma_f32_16x16x32_bf16(
                       false, alo1, false, bhi, (short)0, acc1, false, false);
            acc2 = __builtin_amdgcn_wmma_f32_16x16x32_bf16(
                       false, alo2, false, bhi, (short)0, acc2, false, false);

            #pragma unroll
            for (int r = 0; r < 8; ++r) {
                bool g1 = acc1[r] > bv1[r];           // branchless cndmask
                bi1[r] = g1 ? n : bi1[r];
                bv1[r] = g1 ? acc1[r] : bv1[r];
                bool g2 = acc2[r] > bv2[r];
                bi2[r] = g2 ? n : bi2[r];
                bv2[r] = g2 ? acc2[r] : bv2[r];
            }
        }

        // ---- argmax across columns: transpose keys through LDS, then each
        //      lane reduces its own point's 16 keys in-register.
        //      key: high = ordered score, low = ~idx (ties -> smaller idx)
        //      D layout: slot r = point r (lanes 0-15) / r+8 (lanes 16-31) ---
        #pragma unroll
        for (int r = 0; r < 8; ++r) {
            int pt1 = 8 * grp + r;
            wk[pt1 * KEY_ROW + l16] =
                ((unsigned long long)f32_ordered(bv1[r]) << 32)
                | (unsigned int)(~bi1[r]);
            wk[(16 + pt1) * KEY_ROW + l16] =
                ((unsigned long long)f32_ordered(bv2[r]) << 32)
                | (unsigned int)(~bi2[r]);
        }
        // lane L owns point L of this 32-point tile (same-wave LDS, in order)
        unsigned long long best = wk[lane * KEY_ROW + 0];
        #pragma unroll
        for (int t = 1; t < 16; ++t) {
            unsigned long long k = wk[lane * KEY_ROW + t];
            best = (k > best) ? k : best;             // branchless u64 max
        }
        int myIdx = (int)(~(unsigned int)best);

        if (!OUTPUT_MODE) {
            if (tile * PTS_PER_WAVE + lane < N)
                atomicAdd(&s_cnt[myIdx], 1.0f);
        }
        // redistribute winners to A-layout lanes (lane = point for src)
        int myIdx1 = __shfl(myIdx, l16);
        int myIdx2 = __shfl(myIdx, 16 + l16);

        if (!OUTPUT_MODE) {
            // accumulate this lane's 16 dims of each point into LDS sums
            if (v1) {
                float* dst = s_sums + myIdx1 * DIM + 8 * grp;
                #pragma unroll
                for (int e = 0; e < 8; ++e) atomicAdd(dst + e,      af1[e]);
                #pragma unroll
                for (int e = 0; e < 8; ++e) atomicAdd(dst + 16 + e, af1[8 + e]);
            }
            if (v2) {
                float* dst = s_sums + myIdx2 * DIM + 8 * grp;
                #pragma unroll
                for (int e = 0; e < 8; ++e) atomicAdd(dst + e,      af2[e]);
                #pragma unroll
                for (int e = 0; e < 8; ++e) atomicAdd(dst + 16 + e, af2[8 + e]);
            }
        } else {
            // gather final centroid rows: lane writes its 16 dims per point
            if (v1) {
                const float* src = c + (size_t)myIdx1 * DIM + 8 * grp;
                float* op = out + (size_t)p1 * DIM + 8 * grp;
                *(float4*)(op + 0)  = *(const float4*)(src + 0);
                *(float4*)(op + 4)  = *(const float4*)(src + 4);
                *(float4*)(op + 16) = *(const float4*)(src + 16);
                *(float4*)(op + 20) = *(const float4*)(src + 20);
            }
            if (v2) {
                const float* src = c + (size_t)myIdx2 * DIM + 8 * grp;
                float* op = out + (size_t)p2 * DIM + 8 * grp;
                *(float4*)(op + 0)  = *(const float4*)(src + 0);
                *(float4*)(op + 4)  = *(const float4*)(src + 4);
                *(float4*)(op + 16) = *(const float4*)(src + 16);
                *(float4*)(op + 20) = *(const float4*)(src + 20);
            }
        }
    }

    if (!OUTPUT_MODE) {
        __syncthreads();
        // flush workgroup partials: thread tid owns centroid tid
        float cnt = s_cnt[tid];
        if (cnt != 0.f) unsafeAtomicAdd(&g_cnt[tid], cnt);
        #pragma unroll 4
        for (int d = 0; d < DIM; ++d) {
            float v = s_sums[tid * DIM + d];
            if (v != 0.f) unsafeAtomicAdd(&g_sums[tid * DIM + d], v);
        }
    }
}

// c = cnt>0 ? sums/cnt : 0 ; also re-zero accumulators for the next iteration
__global__ __launch_bounds__(KCENT)
void kmeans_update_kernel(float* __restrict__ c,
                          float* __restrict__ g_sums,
                          float* __restrict__ g_cnt)
{
    int j = threadIdx.x;
    float cnt = g_cnt[j];
    float inv = (cnt > 0.f) ? (1.f / cnt) : 0.f;
    #pragma unroll 4
    for (int d = 0; d < DIM; ++d) {
        float s = g_sums[j * DIM + d];
        c[j * DIM + d] = s * inv;
        g_sums[j * DIM + d] = 0.f;
    }
    g_cnt[j] = 0.f;
}

extern "C" void kernel_launch(void* const* d_in, const int* in_sizes, int n_in,
                              void* d_out, int out_size, void* d_ws, size_t ws_size,
                              hipStream_t stream)
{
    const float* x  = (const float*)d_in[0];
    const float* c0 = (const float*)d_in[1];
    const int N = in_sizes[0] / DIM;              // 1,000,000
    const int numTiles = (N + PTS_PER_WAVE - 1) / PTS_PER_WAVE;

    float* c_cur  = (float*)d_ws;                 // 256*32 f32
    float* g_sums = c_cur + KCENT * DIM;          // 256*32 f32
    float* g_cnt  = g_sums + KCENT * DIM;         // 256 f32

    hipMemcpyAsync(c_cur, c0, KCENT * DIM * sizeof(float),
                   hipMemcpyDeviceToDevice, stream);
    hipMemsetAsync(g_sums, 0, (KCENT * DIM + KCENT) * sizeof(float), stream);

    int grid = 1024;
    if (grid * WAVES_PER_WG > numTiles) grid = (numTiles + WAVES_PER_WG - 1) / WAVES_PER_WG;
    if (grid < 1) grid = 1;

    for (int it = 0; it < 5; ++it) {
        kmeans_assign_kernel<false><<<grid, WG_SIZE, SMEM_BYTES, stream>>>(
            x, c_cur, g_sums, g_cnt, nullptr, N, numTiles);
        kmeans_update_kernel<<<1, KCENT, 0, stream>>>(c_cur, g_sums, g_cnt);
    }
    kmeans_assign_kernel<true><<<grid, WG_SIZE, SMEM_BYTES, stream>>>(
        x, c_cur, nullptr, nullptr, (float*)d_out, N, numTiles);
}
